// STGNNFeatureExtractor_55001351193098
// MI455X (gfx1250) — matrix operand order repaired and checked by reading, not compiled, not run on hardware
//
#include <hip/hip_runtime.h>
#include <hip/hip_bf16.h>

typedef __attribute__((ext_vector_type(16))) _Float16 v16h;
typedef __attribute__((ext_vector_type(8)))  _Float16 v8h;
typedef __attribute__((ext_vector_type(8)))  float    v8f;

#define NEG_SLOPE 0.2f
#define BN_EPS 1e-5f

// ---------------- utility fills ----------------
__global__ void k_fill(float* p, float v, int n) {
    int i = blockIdx.x * blockDim.x + threadIdx.x;
    if (i < n) p[i] = v;
}

// ---------------- layer 1 projection + attention dots ----------------
// h1[n,t] = x[n,0]*W1[0,t] + x[n,1]*W1[1,t]   (t = head*64 + c, 256 wide)
// as1[n,h] = sum_c h1[n,h,c]*a_src[h,c] ; ad1 likewise
__global__ void k_proj1(const float* __restrict__ x, const float* __restrict__ W1,
                        const float* __restrict__ aS, const float* __restrict__ aD,
                        float* __restrict__ h1, float* __restrict__ as1, float* __restrict__ ad1) {
    int n = blockIdx.x, t = threadIdx.x;
    float x0 = x[n * 2 + 0], x1 = x[n * 2 + 1];
    float h = x0 * W1[t] + x1 * W1[256 + t];
    h1[(size_t)n * 256 + t] = h;
    __shared__ float sp[256], sq[256];
    sp[t] = h * aS[t];
    sq[t] = h * aD[t];
    __syncthreads();
    for (int s = 32; s > 0; s >>= 1) {
        if ((t & 63) < s) { sp[t] += sp[t + s]; sq[t] += sq[t + s]; }
        __syncthreads();
    }
    if ((t & 63) == 0) {
        as1[n * 4 + (t >> 6)] = sp[t];
        ad1[n * 4 + (t >> 6)] = sq[t];
    }
}

// ---------------- segment softmax over edges ----------------
__device__ __forceinline__ void atomicMaxFloat(float* a, float v) {
    if (v >= 0.f) atomicMax((int*)a, __float_as_int(v));
    else          atomicMin((unsigned int*)a, __float_as_uint(v));
}

__device__ __forceinline__ void edge_sd(const int* ei, int E, int e, int& s, int& d) {
    if (e < E) { s = ei[e]; d = ei[E + e]; }
    else       { s = d = e - E; }   // self loops appended after real edges
}

__global__ void k_edge_max(const int* __restrict__ ei, int E, int Et, int H,
                           const float* __restrict__ as, const float* __restrict__ ad,
                           float* __restrict__ m) {
    int idx = blockIdx.x * blockDim.x + threadIdx.x;
    if (idx >= Et * H) return;
    int e = idx / H, h = idx - e * H;
    int s, d; edge_sd(ei, E, e, s, d);
    float v = as[s * H + h] + ad[d * H + h];
    v = (v > 0.f) ? v : NEG_SLOPE * v;
    atomicMaxFloat(&m[d * H + h], v);
}

__global__ void k_edge_expsum(const int* __restrict__ ei, int E, int Et, int H,
                              const float* __restrict__ as, const float* __restrict__ ad,
                              const float* __restrict__ m,
                              float* __restrict__ ex, float* __restrict__ den) {
    int idx = blockIdx.x * blockDim.x + threadIdx.x;
    if (idx >= Et * H) return;
    int e = idx / H, h = idx - e * H;
    int s, d; edge_sd(ei, E, e, s, d);
    float v = as[s * H + h] + ad[d * H + h];
    v = (v > 0.f) ? v : NEG_SLOPE * v;
    float xv = __expf(v - m[d * H + h]);
    ex[idx] = xv;
    atomicAdd(&den[d * H + h], xv);
}

// one block per edge, blockDim = F (= H * C), shiftC = log2(C)
__global__ void k_edge_aggregate(const int* __restrict__ ei, int E, int H, int F, int shiftC,
                                 const float* __restrict__ ex, const float* __restrict__ den,
                                 const float* __restrict__ hfeat, float* __restrict__ out) {
    int e = blockIdx.x, t = threadIdx.x;
    int s, d; edge_sd(ei, E, e, s, d);
    __shared__ float al[8];
    if (t < H) al[t] = ex[e * H + t] / (den[d * H + t] + 1e-16f);
    __syncthreads();
    int hh = t >> shiftC;
    atomicAdd(&out[(size_t)d * F + t], al[hh] * hfeat[(size_t)s * F + t]);
}

// ---------------- batch-norm helpers ----------------
__global__ void k_colsum(const float* __restrict__ in, const float* __restrict__ bvec,
                         float* __restrict__ sums, float* __restrict__ sq, int N, int C) {
    int c = threadIdx.x;          // blockDim == C
    int r0 = blockIdx.x * 64;
    int r1 = r0 + 64; if (r1 > N) r1 = N;
    float s = 0.f, s2 = 0.f;
    for (int r = r0; r < r1; ++r) {
        float v = in[(size_t)r * C + c] + bvec[c];
        s += v; s2 += v * v;
    }
    atomicAdd(&sums[c], s);
    atomicAdd(&sq[c], s2);
}

__global__ void k_bn_finalize(const float* sums, const float* sq,
                              const float* gamma, const float* beta,
                              float* scale, float* shift, int N, int C) {
    int c = threadIdx.x; if (c >= C) return;
    float inv = 1.0f / (float)N;
    float mu = sums[c] * inv;
    float var = sq[c] * inv - mu * mu;
    float sc = gamma[c] * rsqrtf(var + BN_EPS);
    scale[c] = sc;
    shift[c] = beta[c] - mu * sc;
}

__device__ __forceinline__ float elu1(float v) { return v > 0.f ? v : (__expf(v) - 1.0f); }

// normalize + ELU, emit f16 (feeds WMMA GEMM)
__global__ void k_bn_elu_f16(const float* __restrict__ in, const float* __restrict__ bvec,
                             const float* __restrict__ scale, const float* __restrict__ shift,
                             _Float16* __restrict__ out, int total, int cmask) {
    int i = blockIdx.x * blockDim.x + threadIdx.x;
    if (i >= total) return;
    int c = i & cmask;
    float v = (in[i] + bvec[c]) * scale[c] + shift[c];
    out[i] = (_Float16)elu1(v);
}

// normalize + ELU, emit f32
__global__ void k_bn_elu_f32(const float* __restrict__ in, const float* __restrict__ bvec,
                             const float* __restrict__ scale, const float* __restrict__ shift,
                             float* __restrict__ out, int total, int cmask) {
    int i = blockIdx.x * blockDim.x + threadIdx.x;
    if (i >= total) return;
    int c = i & cmask;
    float v = (in[i] + bvec[c]) * scale[c] + shift[c];
    out[i] = elu1(v);
}

// ---------------- W2 -> WMMA-ready swizzled f16 B ----------------
// layout index: ((kt*8 + nt)*32 + lane)*16 + j
//   col = nt*16 + (lane&15) ; k = kt*32 + (lane<16 ? j : 16+j)
__global__ void k_cvt_w2(const float* __restrict__ W2, _Float16* __restrict__ B16) {
    int i = blockIdx.x * blockDim.x + threadIdx.x;
    if (i >= 8 * 8 * 32 * 16) return;
    int j = i & 15;
    int lane = (i >> 4) & 31;
    int nt = (i >> 9) & 7;
    int kt = i >> 12;
    int col = nt * 16 + (lane & 15);
    int k = kt * 32 + ((lane < 16) ? j : 16 + j);
    B16[i] = (_Float16)W2[k * 128 + col];
}

// ---------------- layer-2 GEMM via V_WMMA_F32_16X16X32_F16 ----------------
// A: [N,256] f16 row-major; B16: pre-swizzled W2; out h2: [N,128] f32
// 1 block = 16 rows; 8 waves per block = the 8 column tiles; K loop = 8 steps of 32
__global__ void k_gemm2(const _Float16* __restrict__ A, const _Float16* __restrict__ B16,
                        float* __restrict__ h2) {
    int wave = threadIdx.x >> 5;   // 0..7 -> N tile
    int lane = threadIdx.x & 31;
    int m0 = blockIdx.x * 16;
    int row = m0 + (lane & 15);
    const _Float16* arow = A + (size_t)row * 256;
    int koff = (lane < 16) ? 0 : 16;
    v8f acc = {};
    for (int kt = 0; kt < 8; ++kt) {
        int k0 = kt * 32;
        v8h lo = *(const v8h*)(arow + k0 + koff);
        v8h hi = *(const v8h*)(arow + k0 + koff + 8);
        v16h a = __builtin_shufflevector(lo, hi, 0, 1, 2, 3, 4, 5, 6, 7,
                                                  8, 9, 10, 11, 12, 13, 14, 15);
        v16h b = *(const v16h*)(B16 + ((size_t)(kt * 8 + wave) * 32 + lane) * 16);
        acc = __builtin_amdgcn_wmma_f32_16x16x32_f16(
            /*neg_a=*/false, a, /*neg_b=*/false, b,
            /*c_mod=*/(short)0, acc, /*reuse_a=*/false, /*reuse_b=*/false);
    }
    int col = wave * 16 + (lane & 15);
    int rbase = m0 + ((lane < 16) ? 0 : 8);
    #pragma unroll
    for (int i = 0; i < 8; ++i)
        h2[(size_t)(rbase + i) * 128 + col] = acc[i];
}

// ---------------- layer-2 attention dots ----------------
__global__ void k_attdot2(const float* __restrict__ h2,
                          const float* __restrict__ aS, const float* __restrict__ aD,
                          float* __restrict__ as2, float* __restrict__ ad2) {
    int n = blockIdx.x, t = threadIdx.x;  // blockDim = 128
    __shared__ float sp[128], sq[128];
    float v = h2[(size_t)n * 128 + t];
    sp[t] = v * aS[t];
    sq[t] = v * aD[t];
    __syncthreads();
    for (int s = 64; s > 0; s >>= 1) {
        if (t < s) { sp[t] += sp[t + s]; sq[t] += sq[t + s]; }
        __syncthreads();
    }
    if (t == 0) { as2[n] = sp[0]; ad2[n] = sq[0]; }
}

// ---------------- global mean pool ----------------
__global__ void k_counts(const int* __restrict__ batch, float* __restrict__ counts, int N) {
    int i = blockIdx.x * blockDim.x + threadIdx.x;
    if (i < N) atomicAdd(&counts[batch[i]], 1.0f);
}
__global__ void k_pool_sum(const float* __restrict__ hbn2, const int* __restrict__ batch,
                           float* __restrict__ out, int N) {
    int i = blockIdx.x * blockDim.x + threadIdx.x;
    if (i >= N * 128) return;
    int n = i >> 7, c = i & 127;
    atomicAdd(&out[batch[n] * 128 + c], hbn2[i]);
}
__global__ void k_pool_div(float* __restrict__ out, const float* __restrict__ counts, int total) {
    int i = blockIdx.x * blockDim.x + threadIdx.x;
    if (i >= total) return;
    out[i] = out[i] / fmaxf(counts[i >> 7], 1.0f);
}

// =====================================================================
extern "C" void kernel_launch(void* const* d_in, const int* in_sizes, int n_in,
                              void* d_out, int out_size, void* d_ws, size_t ws_size,
                              hipStream_t stream) {
    const float* x   = (const float*)d_in[0];
    const int* ei    = (const int*)d_in[1];
    const int* batch = (const int*)d_in[2];
    const float* W1  = (const float*)d_in[4];
    const float* aS1 = (const float*)d_in[5];
    const float* aD1 = (const float*)d_in[6];
    const float* b1  = (const float*)d_in[7];
    const float* g1  = (const float*)d_in[8];
    const float* be1 = (const float*)d_in[9];
    const float* W2  = (const float*)d_in[10];
    const float* aS2 = (const float*)d_in[11];
    const float* aD2 = (const float*)d_in[12];
    const float* b2  = (const float*)d_in[13];
    const float* g2  = (const float*)d_in[14];
    const float* be2 = (const float*)d_in[15];

    const int N  = in_sizes[0] / 2;   // 50000 (divisible by 16)
    const int E  = in_sizes[1] / 2;   // 320000
    const int Et = E + N;             // edges + self loops
    const int G  = out_size / 128;    // 128 graphs
    const int F1 = 256, F2 = 128, H1 = 4;

    char* w = (char*)d_ws;
    size_t o = 0;
    auto take = [&](size_t bytes) -> char* {
        char* p = w + o;
        o = (o + bytes + 255) & ~(size_t)255;
        return p;
    };
    // big aliased regions
    char* bufA = take((size_t)N * F1 * 4);   // h1 -> {A16, out2}
    char* bufB = take((size_t)N * F1 * 4);   // out1 -> {h2, hbn2}
    float*    h1   = (float*)bufA;
    _Float16* A16  = (_Float16*)bufA;                      // N*256 f16 = first half
    float*    out2 = (float*)(bufA + (size_t)N * F1 * 2);  // N*128 f32 = second half
    float*    out1 = (float*)bufB;
    float*    h2   = (float*)bufB;                          // N*128 f32 = first half
    float*    hbn2 = (float*)(bufB + (size_t)N * F2 * 4);   // N*128 f32 = second half
    // small buffers
    float* as1  = (float*)take((size_t)N * H1 * 4);
    float* ad1  = (float*)take((size_t)N * H1 * 4);
    float* m1   = (float*)take((size_t)N * H1 * 4);
    float* den1 = (float*)take((size_t)N * H1 * 4);
    float* ex1  = (float*)take((size_t)Et * H1 * 4);
    float* as2  = (float*)take((size_t)N * 4);
    float* ad2  = (float*)take((size_t)N * 4);
    float* m2   = (float*)take((size_t)N * 4);
    float* den2 = (float*)take((size_t)N * 4);
    float* ex2  = (float*)take((size_t)Et * 4);
    _Float16* B16 = (_Float16*)take((size_t)256 * 128 * 2);
    float* sums1  = (float*)take(256 * 4);
    float* sq1    = (float*)take(256 * 4);
    float* scale1 = (float*)take(256 * 4);
    float* shift1 = (float*)take(256 * 4);
    float* sums2  = (float*)take(128 * 4);
    float* sq2    = (float*)take(128 * 4);
    float* scale2 = (float*)take(128 * 4);
    float* shift2 = (float*)take(128 * 4);
    float* counts = (float*)take((size_t)G * 4);

    const float NEG_INF = -__builtin_huge_valf();
    auto blks = [](int n, int t) { return (n + t - 1) / t; };

    // ---- init layer-1 accumulators + stats + output ----
    k_fill<<<blks(N * H1, 256), 256, 0, stream>>>(m1, NEG_INF, N * H1);
    k_fill<<<blks(N * H1, 256), 256, 0, stream>>>(den1, 0.f, N * H1);
    k_fill<<<blks(N * F1, 256), 256, 0, stream>>>(out1, 0.f, N * F1);
    k_fill<<<1, 256, 0, stream>>>(sums1, 0.f, 256);
    k_fill<<<1, 256, 0, stream>>>(sq1, 0.f, 256);
    k_fill<<<1, 128, 0, stream>>>(sums2, 0.f, 128);
    k_fill<<<1, 128, 0, stream>>>(sq2, 0.f, 128);
    k_fill<<<1, 256, 0, stream>>>(counts, 0.f, G);
    k_fill<<<blks(out_size, 256), 256, 0, stream>>>((float*)d_out, 0.f, out_size);

    // ---- layer 1 ----
    k_proj1<<<N, 256, 0, stream>>>(x, W1, aS1, aD1, h1, as1, ad1);
    k_edge_max<<<blks(Et * H1, 256), 256, 0, stream>>>(ei, E, Et, H1, as1, ad1, m1);
    k_edge_expsum<<<blks(Et * H1, 256), 256, 0, stream>>>(ei, E, Et, H1, as1, ad1, m1, ex1, den1);
    k_edge_aggregate<<<Et, 256, 0, stream>>>(ei, E, H1, F1, 6, ex1, den1, h1, out1);

    // ---- BN1 + ELU -> f16 GEMM input ----
    k_colsum<<<blks(N, 64), 256, 0, stream>>>(out1, b1, sums1, sq1, N, F1);
    k_bn_finalize<<<1, 256, 0, stream>>>(sums1, sq1, g1, be1, scale1, shift1, N, F1);
    k_bn_elu_f16<<<blks(N * F1, 256), 256, 0, stream>>>(out1, b1, scale1, shift1, A16,
                                                        N * F1, F1 - 1);

    // ---- layer-2 GEMM (WMMA) ----
    k_cvt_w2<<<blks(8 * 8 * 32 * 16, 256), 256, 0, stream>>>(W2, B16);
    k_gemm2<<<N / 16, 256, 0, stream>>>(A16, B16, h2);

    // ---- layer-2 attention / softmax / aggregate ----
    k_fill<<<blks(N, 256), 256, 0, stream>>>(m2, NEG_INF, N);
    k_fill<<<blks(N, 256), 256, 0, stream>>>(den2, 0.f, N);
    k_fill<<<blks(N * F2, 256), 256, 0, stream>>>(out2, 0.f, N * F2);
    k_attdot2<<<N, 128, 0, stream>>>(h2, aS2, aD2, as2, ad2);
    k_edge_max<<<blks(Et, 256), 256, 0, stream>>>(ei, E, Et, 1, as2, ad2, m2);
    k_edge_expsum<<<blks(Et, 256), 256, 0, stream>>>(ei, E, Et, 1, as2, ad2, m2, ex2, den2);
    k_edge_aggregate<<<Et, 128, 0, stream>>>(ei, E, 1, F2, 7, ex2, den2, h2, out2);

    // ---- BN2 + ELU ----
    k_colsum<<<blks(N, 64), 128, 0, stream>>>(out2, b2, sums2, sq2, N, F2);
    k_bn_finalize<<<1, 128, 0, stream>>>(sums2, sq2, g2, be2, scale2, shift2, N, F2);
    k_bn_elu_f32<<<blks(N * F2, 256), 256, 0, stream>>>(out2, b2, scale2, shift2, hbn2,
                                                        N * F2, F2 - 1);

    // ---- global mean pool ----
    k_counts<<<blks(N, 256), 256, 0, stream>>>(batch, counts, N);
    k_pool_sum<<<blks(N * F2, 256), 256, 0, stream>>>(hbn2, batch, (float*)d_out, N);
    k_pool_div<<<blks(out_size, 256), 256, 0, stream>>>((float*)d_out, counts, out_size);
}
